// BiDirectionalFusionModule_28887950032927
// MI455X (gfx1250) — matrix operand order repaired and checked by reading, not compiled, not run on hardware
//
#include <hip/hip_runtime.h>

// MI455X / gfx1250: wave32, WMMA 16x16x32 f16->f32
typedef __attribute__((ext_vector_type(16))) _Float16 v16h;
typedef __attribute__((ext_vector_type(8)))  _Float16 v8h;
typedef __attribute__((ext_vector_type(8)))  float    v8f;
typedef int vsi4 __attribute__((vector_size(16)));   // matches async-builtin pointee

#define NBATCH 4
#define NCH    256
#define NTOK   4096
#define CKDIM  32
#define BSTR   40   // B-tile row stride in halves: 80B = 16B-aligned, low bank conflict

// ---- gfx1250 async global->LDS copy (guarded; falls back to VGPR copy) ----
#if __has_builtin(__builtin_amdgcn_global_load_async_to_lds_b128) && \
    __has_builtin(__builtin_amdgcn_s_wait_asynccnt)
#define HAVE_ASYNC_LDS 1
#else
#define HAVE_ASYNC_LDS 0
#endif

__device__ __forceinline__ void copy16_g2lds(const _Float16* g, _Float16* l) {
#if HAVE_ASYNC_LDS
  __builtin_amdgcn_global_load_async_to_lds_b128(
      (__attribute__((address_space(1))) vsi4*)(uintptr_t)g,
      (__attribute__((address_space(3))) vsi4*)(vsi4*)(void*)l, 0, 0);
#else
  *(v8h*)l = *(const v8h*)g;
#endif
}

__device__ __forceinline__ void async_join() {
#if HAVE_ASYNC_LDS
  __builtin_amdgcn_s_wait_asynccnt(0);
#endif
}

__device__ __forceinline__ v8f wmma_f16(v16h a, v16h b, v8f c) {
  // D = A(16x32 f16) * B(32x16 f16) + C(16x16 f32)
  return __builtin_amdgcn_wmma_f32_16x16x32_f16(false, a, false, b, (short)0, c,
                                                false, false);
}

__device__ __forceinline__ v8f vzero8() {
  v8f z;
#pragma unroll
  for (int e = 0; e < 8; ++e) z[e] = 0.0f;
  return z;
}

__device__ __forceinline__ v8h hzero8() {
  v8h z;
#pragma unroll
  for (int e = 0; e < 8; ++e) z[e] = (_Float16)0.0f;
  return z;
}

// A-operand fragment: 16(M) x 32(K) f16 tile, LDS row-major [16][rs], rs mult of 8.
// ISA 7.12.2: lane L -> M=L&15; K runs {kb..kb+7} and {16+kb..16+kb+7}, kb=(L&16)?8:0.
__device__ __forceinline__ v16h load_a_frag(const _Float16* t, int rs) {
  int lane = threadIdx.x & 31;
  const _Float16* p = t + (lane & 15) * rs + ((lane & 16) ? 8 : 0);
  v8h lo = *(const v8h*)p;         // K = kb+0 .. kb+7   -> elements 0..7
  v8h hi = *(const v8h*)(p + 16);  // K = 16+kb .. 16+kb+7 -> elements 8..15
  return __builtin_shufflevector(lo, hi, 0, 1, 2, 3, 4, 5, 6, 7,
                                 8, 9, 10, 11, 12, 13, 14, 15);
}

// B-operand fragment: 32(K) x 16(N) f16. Tile stored TRANSPOSED in LDS as [N][K]
// with row stride rs. Lane L -> N = n0+(L&15); K runs {kb..kb+7} and {16+kb..16+kb+7}, kb=(L&16)?8:0.
__device__ __forceinline__ v16h load_b_frag(const _Float16* t, int rs, int n0) {
  int lane = threadIdx.x & 31;
  const _Float16* p = t + (n0 + (lane & 15)) * rs + ((lane & 16) ? 8 : 0);
  v8h lo = *(const v8h*)p;        // K = kb+0..7
  v8h hi = *(const v8h*)(p + 16); // K = 16+kb..16+kb+7
  return __builtin_shufflevector(lo, hi, 0, 1, 2, 3, 4, 5, 6, 7,
                                 8, 9, 10, 11, 12, 13, 14, 15);
}

// ---------------------------------------------------------------------------
// One-time weight repack: wf (256,512,3,3) f32 -> wp[tap][cout][cin] f16.
// ---------------------------------------------------------------------------
__global__ __launch_bounds__(256)
void pack_wf_kernel(const float* __restrict__ wf, _Float16* __restrict__ wp) {
  int idx = blockIdx.x * 256 + threadIdx.x;    // over 9*256*512
  int i    = idx & 511;
  int rest = idx >> 9;
  int o    = rest & 255;
  int tap  = rest >> 8;
  wp[idx] = (_Float16)wf[(size_t)(o * 512 + i) * 9 + tap];
}

// ---------------------------------------------------------------------------
// Projection GEMM: Y(MxN) = W(MxC) * X(CxN) + bias, output f16.
// transposed=1 -> Y token-major (n*M + m); else channel-major (m*N + n).
// ---------------------------------------------------------------------------
__global__ __launch_bounds__(128)
void proj_gemm_kernel(const float* __restrict__ X, const float* __restrict__ W,
                      const float* __restrict__ bias, _Float16* __restrict__ Y,
                      int M, int transposed) {
  __shared__ alignas(16) _Float16 Wt[16 * 32];     // A tile [M][K]
  __shared__ alignas(16) _Float16 Xt[256 * BSTR];  // B tile [N][K]
  int nt     = blockIdx.x & 15;
  int rest   = blockIdx.x >> 4;
  int mtiles = M >> 4;
  int mt = rest % mtiles;
  int b  = rest / mtiles;
  int m0 = mt * 16;
  int n0 = nt * 256;
  int tid = threadIdx.x, wv = tid >> 5, lane = tid & 31;
  const float* Xb = X + (size_t)b * NCH * NTOK;

  v8f acc[4];
#pragma unroll
  for (int i = 0; i < 4; ++i) acc[i] = vzero8();

  for (int kc = 0; kc < NCH; kc += 32) {
    __syncthreads();
    {  // A tile: 128 float4 quads, one per thread (f32 -> f16 convert)
      int r = tid >> 3, k4 = (tid & 7) * 4;
      float4 f = *(const float4*)&W[(size_t)(m0 + r) * NCH + kc + k4];
      _Float16* d = &Wt[r * 32 + k4];
      d[0] = (_Float16)f.x; d[1] = (_Float16)f.y;
      d[2] = (_Float16)f.z; d[3] = (_Float16)f.w;
    }
    // B tile: transpose f32[K][N] -> f16[N][K]; float4 along n (coalesced)
#pragma unroll
    for (int q = 0; q < 16; ++q) {
      int id = tid + q * 128;          // 0..2047
      int k = id >> 6, n4 = (id & 63) * 4;
      float4 f = *(const float4*)&Xb[(size_t)(kc + k) * NTOK + n0 + n4];
      _Float16* d = &Xt[n4 * BSTR + k];
      d[0]        = (_Float16)f.x;
      d[BSTR]     = (_Float16)f.y;
      d[2 * BSTR] = (_Float16)f.z;
      d[3 * BSTR] = (_Float16)f.w;
    }
    __syncthreads();
    v16h a = load_a_frag(Wt, 32);
#pragma unroll
    for (int bt = 0; bt < 4; ++bt) {
      v16h bb = load_b_frag(Xt, BSTR, wv * 64 + bt * 16);
      acc[bt] = wmma_f16(a, bb, acc[bt]);
    }
  }

  int hb = (lane & 16) ? 8 : 0;
#pragma unroll
  for (int bt = 0; bt < 4; ++bt) {
    int n = n0 + wv * 64 + bt * 16 + (lane & 15);
    if (transposed) {  // 8 contiguous m per lane -> one v8h store
      v8h pack;
#pragma unroll
      for (int v = 0; v < 8; ++v)
        pack[v] = (_Float16)(acc[bt][v] + bias[m0 + hb + v]);
      *(v8h*)&Y[((size_t)b * NTOK + n) * M + m0 + hb] = pack;
    } else {
#pragma unroll
      for (int v = 0; v < 8; ++v) {
        int m = m0 + v + hb;
        Y[((size_t)b * M + m) * NTOK + n] = (_Float16)(acc[bt][v] + bias[m]);
      }
    }
  }
}

// ---------------------------------------------------------------------------
// Flash attention: enhC[b, coff+c, i] = f16(base[b,c,i] + gamma * O[i,c]/l)
// Qt,Kt: (B,N,32) token-major f16; Vch: (B,256,N) channel-major f16.
// enhC: (B,512,N) f16 channel-major (concat buffer; d2r coff=0, r2d coff=256).
// Block: 4 waves, 32 query rows. Wave (ig,ch): rows 16*ig..+15, C half ch.
// ---------------------------------------------------------------------------
__global__ __launch_bounds__(128)
void flash_attn_kernel(const _Float16* __restrict__ Qt, const _Float16* __restrict__ Kt,
                       const _Float16* __restrict__ Vch, const float* __restrict__ base,
                       const float* __restrict__ gamma, _Float16* __restrict__ enhC,
                       int coff) {
  __shared__ alignas(16) _Float16 Qs[32 * 32];        // A tile [i][c]
  __shared__ alignas(16) _Float16 Ks[32 * BSTR];      // B tile [j][c] (N=j, K=c)
  __shared__ alignas(16) _Float16 Vs[256 * BSTR];     // B tile [c][j] (N=c, K=j)
  __shared__ alignas(16) _Float16 Ps[2][16 * 32];     // A tile [i][j]
  int it = blockIdx.x & 127;
  int b  = blockIdx.x >> 7;
  int i0 = it * 32;
  int tid = threadIdx.x, wv = tid >> 5, lane = tid & 31;
  int ig = wv >> 1;  // query-row group
  int ch = wv & 1;   // C half

  {  // Q tile: 128 16B chunks, one per thread (async global->LDS when available)
    int i = tid >> 2, cq = (tid & 3) * 8;
    copy16_g2lds(&Qt[((size_t)b * NTOK + i0 + i) * CKDIM + cq], &Qs[i * 32 + cq]);
  }
  async_join();
  __syncthreads();
  v16h aq = load_a_frag(Qs + ig * 16 * 32, 32);

  float m_run[8], l_run[8];
#pragma unroll
  for (int v = 0; v < 8; ++v) { m_run[v] = -1e30f; l_run[v] = 0.0f; }
  v8f acc[8];
#pragma unroll
  for (int i = 0; i < 8; ++i) acc[i] = vzero8();

  for (int jt = 0; jt < NTOK; jt += 32) {
    __syncthreads();
    {  // K tile: 128 16B chunks, one per thread
      int j = tid >> 2, cq = (tid & 3) * 8;
      copy16_g2lds(&Kt[((size_t)b * NTOK + jt + j) * CKDIM + cq], &Ks[j * BSTR + cq]);
    }
    // V tile: 512 16B chunks, 4 per thread (rows contiguous from channel-major)
#pragma unroll
    for (int q = 0; q < 4; ++q) {
      int id = tid + q * 128;
      int c = id >> 2, jq = (id & 3) * 8;
      copy16_g2lds(&Vch[((size_t)b * NCH + c) * NTOK + jt + jq], &Vs[c * BSTR + jq]);
    }
    async_join();
    __syncthreads();

    // prefetch next tile toward the WGP while this tile computes
    if (jt + 32 < NTOK) {
      int nj = jt + 32;
      const _Float16* pv = Vch + ((size_t)b * NCH + tid * 2) * NTOK + nj;
      __builtin_prefetch(pv, 0, 3);
      __builtin_prefetch(pv + NTOK, 0, 3);
      const _Float16* pk = Kt + ((size_t)b * NTOK + nj + (tid & 31)) * CKDIM;
      __builtin_prefetch(pk, 0, 3);
    }

    v16h bk0 = load_b_frag(Ks, BSTR, 0);
    v16h bk1 = load_b_frag(Ks, BSTR, 16);
    v8f s0 = wmma_f16(aq, bk0, vzero8());
    v8f s1 = wmma_f16(aq, bk1, vzero8());

    // online softmax: each 16-lane half of D-layout holds one full row (j across lanes)
    float scale[8];
#pragma unroll
    for (int v = 0; v < 8; ++v) {
      float mx = fmaxf(s0[v], s1[v]);
#pragma unroll
      for (int off = 1; off < 16; off <<= 1)
        mx = fmaxf(mx, __shfl_xor(mx, off, 32));
      float mnew = fmaxf(m_run[v], mx);
      scale[v] = __expf(m_run[v] - mnew);
      float p0 = __expf(s0[v] - mnew);
      float p1 = __expf(s1[v] - mnew);
      float rs = p0 + p1;
#pragma unroll
      for (int off = 1; off < 16; off <<= 1)
        rs += __shfl_xor(rs, off, 32);
      l_run[v] = l_run[v] * scale[v] + rs;
      m_run[v] = mnew;
      int ir = v + ((lane & 16) ? 8 : 0);
      // both waves of a group write identical values (same deterministic math)
      Ps[ig][ir * 32 + (lane & 15)]      = (_Float16)p0;
      Ps[ig][ir * 32 + 16 + (lane & 15)] = (_Float16)p1;
    }
#pragma unroll
    for (int ct = 0; ct < 8; ++ct)
#pragma unroll
      for (int v = 0; v < 8; ++v) acc[ct][v] *= scale[v];

    v16h af = load_a_frag(Ps[ig], 32);
#pragma unroll
    for (int ct = 0; ct < 8; ++ct) {
      v16h bv = load_b_frag(Vs, BSTR, ch * 128 + ct * 16);
      acc[ct] = wmma_f16(af, bv, acc[ct]);
    }
  }

  float g = gamma[0];
  int hb = (lane & 16) ? 8 : 0;
  float rl[8];
#pragma unroll
  for (int v = 0; v < 8; ++v) rl[v] = 1.0f / l_run[v];
  // Epilogue: 8 elements per lane contiguous in i -> float4 residual reads,
  // one v8h f16 store per tile into the concat buffer.
#pragma unroll
  for (int ct = 0; ct < 8; ++ct) {
    int c = ch * 128 + ct * 16 + (lane & 15);
    size_t rb = ((size_t)b * NCH + c) * NTOK + i0 + ig * 16 + hb;
    float4 b0 = *(const float4*)&base[rb];
    float4 b1 = *(const float4*)&base[rb + 4];
    v8h pack;
    pack[0] = (_Float16)(b0.x + g * acc[ct][0] * rl[0]);
    pack[1] = (_Float16)(b0.y + g * acc[ct][1] * rl[1]);
    pack[2] = (_Float16)(b0.z + g * acc[ct][2] * rl[2]);
    pack[3] = (_Float16)(b0.w + g * acc[ct][3] * rl[3]);
    pack[4] = (_Float16)(b1.x + g * acc[ct][4] * rl[4]);
    pack[5] = (_Float16)(b1.y + g * acc[ct][5] * rl[5]);
    pack[6] = (_Float16)(b1.z + g * acc[ct][6] * rl[6]);
    pack[7] = (_Float16)(b1.w + g * acc[ct][7] * rl[7]);
    *(v8h*)&enhC[((size_t)b * 512 + coff + c) * NTOK + i0 + ig * 16 + hb] = pack;
  }
}

// ---------------------------------------------------------------------------
// Relayout: enhC (B,512,N) channel-major -> enhT (B,N,512) token-major,
// LDS 32x32 tile transpose so both global sides stay coalesced.
// ---------------------------------------------------------------------------
__global__ __launch_bounds__(128)
void relayout_kernel(const _Float16* __restrict__ src, _Float16* __restrict__ dst) {
  __shared__ _Float16 tile[32][33];   // +1 pad: conflict-free transpose
  int nt = blockIdx.x & 127;          // 4096/32
  int ct = (blockIdx.x >> 7) & 15;    // 512/32
  int b  = blockIdx.x >> 11;
  int tid = threadIdx.x;
#pragma unroll
  for (int q = 0; q < 8; ++q) {       // read coalesced along n
    int id = tid + q * 128;
    int c = id >> 5, n = id & 31;
    tile[c][n] = src[((size_t)b * 512 + ct * 32 + c) * NTOK + nt * 32 + n];
  }
  __syncthreads();
#pragma unroll
  for (int q = 0; q < 8; ++q) {       // write coalesced along c
    int id = tid + q * 128;
    int n = id >> 5, c = id & 31;
    dst[((size_t)b * NTOK + nt * 32 + n) * 512 + ct * 32 + c] = tile[c][n];
  }
}

// ---------------------------------------------------------------------------
// Fused 3x3 conv (512->256) + BN + ReLU as 9 shifted 1x1 WMMA GEMMs.
// Inputs: enhT (B,N,512) f16 token-major, wp tap-major f16.
// Block: 4 waves, 64 c_out x 64 tokens (one image row y). Pure 16B staging.
// ---------------------------------------------------------------------------
__global__ __launch_bounds__(128)
void conv_bn_relu_kernel(const _Float16* __restrict__ enhT,
                         const _Float16* __restrict__ wp, const float* __restrict__ bf,
                         const float* __restrict__ bsc, const float* __restrict__ bbi,
                         const float* __restrict__ bmn, const float* __restrict__ bvr,
                         float* __restrict__ out) {
  __shared__ alignas(16) _Float16 Wa[64 * 32];      // A tile [cout][k]
  __shared__ alignas(16) _Float16 Xs[64 * BSTR];    // B tile [x][k]
  int y  = blockIdx.x & 63;
  int t  = blockIdx.x >> 6;
  int mt = t & 3;
  int b  = t >> 2;
  int m0 = mt * 64;
  int tid = threadIdx.x, wv = tid >> 5, lane = tid & 31;

  v8f acc[4];
#pragma unroll
  for (int i = 0; i < 4; ++i) acc[i] = vzero8();

  for (int tap = 0; tap < 9; ++tap) {
    int dy = tap / 3 - 1, dx = tap % 3 - 1;
    int yy = y + dy;
    bool yok = (yy >= 0) && (yy < 64);
    const _Float16* wtap = wp + (size_t)tap * 256 * 512;
    for (int kc = 0; kc < 512; kc += 32) {
      __syncthreads();
      // A tile: 256 16B chunks, 2 per thread
#pragma unroll
      for (int q = 0; q < 2; ++q) {
        int id = tid + q * 128;
        int r = id >> 2, kq = (id & 3) * 8;
        copy16_g2lds(&wtap[(size_t)(m0 + r) * 512 + kc + kq], &Wa[r * 32 + kq]);
      }
      // B tile: 256 16B chunks from token-major concat; zero-fill borders
#pragma unroll
      for (int q = 0; q < 2; ++q) {
        int id = tid + q * 128;
        int x = id >> 2, kq = (id & 3) * 8;
        int xx = x + dx;
        if (yok && xx >= 0 && xx < 64) {
          copy16_g2lds(&enhT[((size_t)b * NTOK + yy * 64 + xx) * 512 + kc + kq],
                       &Xs[x * BSTR + kq]);
        } else {
          *(v8h*)&Xs[x * BSTR + kq] = hzero8();
        }
      }
      async_join();
      __syncthreads();
      v16h a = load_a_frag(Wa + wv * 16 * 32, 32);
#pragma unroll
      for (int bt = 0; bt < 4; ++bt) {
        v16h bb = load_b_frag(Xs, BSTR, bt * 16);
        acc[bt] = wmma_f16(a, bb, acc[bt]);
      }
    }
  }

  int hb = (lane & 16) ? 8 : 0;
#pragma unroll
  for (int bt = 0; bt < 4; ++bt) {
#pragma unroll
    for (int v = 0; v < 8; ++v) {
      int c = m0 + wv * 16 + v + hb;
      int x = bt * 16 + (lane & 15);
      float yv = acc[bt][v] + bf[c];
      float inv = rsqrtf(bvr[c] + 1e-5f);
      yv = (yv - bmn[c]) * (inv * bsc[c]) + bbi[c];
      yv = fmaxf(yv, 0.0f);
      out[((size_t)b * 256 + c) * 4096 + (size_t)y * 64 + x] = yv;
    }
  }
}

// ---------------------------------------------------------------------------
extern "C" void kernel_launch(void* const* d_in, const int* in_sizes, int n_in,
                              void* d_out, int out_size, void* d_ws, size_t ws_size,
                              hipStream_t stream) {
  (void)in_sizes; (void)n_in; (void)out_size; (void)ws_size;
  const float* f_rgb     = (const float*)d_in[0];
  const float* f_depth   = (const float*)d_in[1];
  const float* wq_d2r    = (const float*)d_in[2];
  const float* bq_d2r    = (const float*)d_in[3];
  const float* wk_d2r    = (const float*)d_in[4];
  const float* bk_d2r    = (const float*)d_in[5];
  const float* wv_d2r    = (const float*)d_in[6];
  const float* bv_d2r    = (const float*)d_in[7];
  const float* gamma_d2r = (const float*)d_in[8];
  const float* wq_r2d    = (const float*)d_in[9];
  const float* bq_r2d    = (const float*)d_in[10];
  const float* wk_r2d    = (const float*)d_in[11];
  const float* bk_r2d    = (const float*)d_in[12];
  const float* wv_r2d    = (const float*)d_in[13];
  const float* bv_r2d    = (const float*)d_in[14];
  const float* gamma_r2d = (const float*)d_in[15];
  const float* wf        = (const float*)d_in[16];
  const float* bf        = (const float*)d_in[17];
  const float* bn_scale  = (const float*)d_in[18];
  const float* bn_bias   = (const float*)d_in[19];
  const float* bn_mean   = (const float*)d_in[20];
  const float* bn_var    = (const float*)d_in[21];

  char* ws = (char*)d_ws;
  size_t off = 0;
  auto salloc = [&](size_t bytes) -> void* {
    void* p = ws + off;
    off += (bytes + 255) & ~(size_t)255;
    return p;
  };
  const size_t qk_elems = (size_t)NBATCH * NTOK * CKDIM;   // 0.5M
  const size_t v_elems  = (size_t)NBATCH * NTOK * NCH;     // 4M
  const size_t eh_elems = (size_t)NBATCH * NTOK * 512;     // 8M (concat)
  const size_t wp_elems = (size_t)9 * 256 * 512;           // 1.125M
  _Float16* Qt_a  = (_Float16*)salloc(qk_elems * 2);
  _Float16* Kt_a  = (_Float16*)salloc(qk_elems * 2);
  _Float16* Vc_a  = (_Float16*)salloc(v_elems * 2);
  _Float16* Qt_b  = (_Float16*)salloc(qk_elems * 2);
  _Float16* Kt_b  = (_Float16*)salloc(qk_elems * 2);
  _Float16* Vc_b  = (_Float16*)salloc(v_elems * 2);
  _Float16* enhC  = (_Float16*)salloc(eh_elems * 2);
  _Float16* enhT  = (_Float16*)salloc(eh_elems * 2);
  _Float16* wpack = (_Float16*)salloc(wp_elems * 2);

  dim3 blk(128);
  // One-time conv-weight repack to tap-major f16
  pack_wf_kernel<<<dim3(wp_elems / 256), dim3(256), 0, stream>>>(wf, wpack);
  // Projections (d2r: Q on rgb, K/V on depth; r2d: Q on depth, K/V on rgb)
  proj_gemm_kernel<<<dim3(NBATCH * 2 * 16), blk, 0, stream>>>(f_rgb,   wq_d2r, bq_d2r, Qt_a, 32, 1);
  proj_gemm_kernel<<<dim3(NBATCH * 2 * 16), blk, 0, stream>>>(f_depth, wk_d2r, bk_d2r, Kt_a, 32, 1);
  proj_gemm_kernel<<<dim3(NBATCH * 16 * 16), blk, 0, stream>>>(f_depth, wv_d2r, bv_d2r, Vc_a, 256, 0);
  proj_gemm_kernel<<<dim3(NBATCH * 2 * 16), blk, 0, stream>>>(f_depth, wq_r2d, bq_r2d, Qt_b, 32, 1);
  proj_gemm_kernel<<<dim3(NBATCH * 2 * 16), blk, 0, stream>>>(f_rgb,   wk_r2d, bk_r2d, Kt_b, 32, 1);
  proj_gemm_kernel<<<dim3(NBATCH * 16 * 16), blk, 0, stream>>>(f_rgb,   wv_r2d, bv_r2d, Vc_b, 256, 0);
  // Flash attention + residual -> f16 concat (channel-major), no NxN materialization
  flash_attn_kernel<<<dim3(NBATCH * 128), blk, 0, stream>>>(Qt_a, Kt_a, Vc_a, f_rgb,
                                                            gamma_d2r, enhC, 0);
  flash_attn_kernel<<<dim3(NBATCH * 128), blk, 0, stream>>>(Qt_b, Kt_b, Vc_b, f_depth,
                                                            gamma_r2d, enhC, 256);
  // Channel-major -> token-major for pure-16B conv staging
  relayout_kernel<<<dim3(NBATCH * 16 * 128), blk, 0, stream>>>(enhC, enhT);
  // Fused 3x3 conv + BN + ReLU
  conv_bn_relu_kernel<<<dim3(NBATCH * 4 * 64), blk, 0, stream>>>(
      enhT, wpack, bf, bn_scale, bn_bias, bn_mean, bn_var, (float*)d_out);
}
